// BalancedGIFNeuron_16965120819977
// MI455X (gfx1250) — matrix operand order repaired and checked by reading, not compiled, not run on hardware
//
#include <hip/hip_runtime.h>
#include <hip/hip_bf16.h>
#include <math.h>

// ---------------------------------------------------------------------------
// BalancedGIFNeuron for MI455X (gfx1250, wave32, WMMA)
//
// Phase 1: f32 -> bf16 conversion of x and [W_exc;W_inh] into workspace.
// Phase 2: WMMA bf16 GEMM (M=16384, N=2048, K=1024) with fused bias + ReLU +
//          Dale sign epilogue, producing i_all (f32) in workspace.
// Phase 3: per-(batch,neuron) sequential scan over T=1024, writing spikes and
//          final (v, theta) into d_out.
// ---------------------------------------------------------------------------

typedef float  v8f   __attribute__((ext_vector_type(8)));
typedef __bf16 v16bf __attribute__((ext_vector_type(16)));
typedef __bf16 bf16x4 __attribute__((ext_vector_type(4)));

// ---- problem constants -----------------------------------------------------
#define BATCH      16
#define TSTEPS     1024
#define KDIM       1024      // INPUT_DIM
#define NDIM       2048      // HIDDEN_DIM
#define MDIM       (BATCH * TSTEPS)   // 16384
#define EXC_DIM    1638
#define L_LEVELS   16.0f
#define ALPHA_C    0.01f
#define DECAY_C    0.9048374180359595f   // exp(-1/10)

// LDS tile row stride (elements). 40 bf16 = 80 B = 5*16 B : keeps 16-byte
// alignment for b128 loads and gives conflict-free banks (gcd(20,64)=4 pattern
// distinct over 16 rows).
#define LDS_STRIDE 40

// ---------------------------------------------------------------------------
// f32 -> bf16 converters
// ---------------------------------------------------------------------------
__global__ void cvt_x_bf16(const float* __restrict__ src,
                           __bf16* __restrict__ dst, int n4) {
    int i = blockIdx.x * blockDim.x + threadIdx.x;
    if (i >= n4) return;
    float4 f = reinterpret_cast<const float4*>(src)[i];
    bf16x4 o;
    o[0] = (__bf16)f.x; o[1] = (__bf16)f.y; o[2] = (__bf16)f.z; o[3] = (__bf16)f.w;
    reinterpret_cast<bf16x4*>(dst)[i] = o;
}

// Pack W_exc (1638 x 1024) and W_inh (410 x 1024) into Wb (2048 x 1024) bf16.
__global__ void cvt_w_bf16(const float* __restrict__ w_exc,
                           const float* __restrict__ w_inh,
                           __bf16* __restrict__ dst, int n4) {
    int i = blockIdx.x * blockDim.x + threadIdx.x;
    if (i >= n4) return;
    int row = i >> 8;                 // (4*i)/1024
    int col4 = i & 255;               // float4 index within row
    const float* src = (row < EXC_DIM)
        ? (w_exc + (size_t)row * KDIM)
        : (w_inh + (size_t)(row - EXC_DIM) * KDIM);
    float4 f = reinterpret_cast<const float4*>(src)[col4];
    bf16x4 o;
    o[0] = (__bf16)f.x; o[1] = (__bf16)f.y; o[2] = (__bf16)f.z; o[3] = (__bf16)f.w;
    reinterpret_cast<bf16x4*>(dst)[i] = o;
}

__global__ void pack_bias(const float* __restrict__ b_exc,
                          const float* __restrict__ b_inh,
                          float* __restrict__ dst) {
    int c = blockIdx.x * blockDim.x + threadIdx.x;
    if (c >= NDIM) return;
    dst[c] = (c < EXC_DIM) ? b_exc[c] : b_inh[c - EXC_DIM];
}

// ---------------------------------------------------------------------------
// WMMA GEMM: i_all[m, n] = dale(relu(X[m, :] . W[n, :] + bias[n]))
//   X : (16384, 1024) bf16 row-major   (A matrix)
//   W : (2048, 1024)  bf16 row-major   (B matrix: B[k][n] = W[n][k])
// Block: 256 threads (8 waves as 4(M) x 2(N)), block tile 128x128, K-step 32.
// Per wave: 32x64 subtile = 2x4 v_wmma_f32_16x16x32_bf16 accumulators.
// ---------------------------------------------------------------------------
__global__ __launch_bounds__(256)
void gemm_bias_dale(const __bf16* __restrict__ X,
                    const __bf16* __restrict__ W,
                    const float* __restrict__ bias,
                    float* __restrict__ out) {
    __shared__ __bf16 As[128 * LDS_STRIDE];   // 128 rows (M) x 32 k
    __shared__ __bf16 Bs[128 * LDS_STRIDE];   // 128 rows (N) x 32 k

    const int tid  = threadIdx.x;
    const int lane = tid & 31;
    const int wave = tid >> 5;
    const int wm   = wave >> 1;        // 0..3 -> M offset wm*32
    const int wn   = wave & 1;         // 0..1 -> N offset wn*64

    const int bm = blockIdx.x;         // 0..127
    const int bn = blockIdx.y;         // 0..15

    // global load assignment: thread -> (row = tid/2, k-half = (tid&1)*16)
    const int lr = tid >> 1;
    const int lh = (tid & 1) << 4;
    const __bf16* xg = X + (size_t)(bm * 128 + lr) * KDIM + lh;
    const __bf16* wg = W + (size_t)(bn * 128 + lr) * KDIM + lh;
    __bf16* asw = As + lr * LDS_STRIDE + lh;
    __bf16* bsw = Bs + lr * LDS_STRIDE + lh;

    v8f acc[2][4];
#pragma unroll
    for (int i = 0; i < 2; ++i)
#pragma unroll
        for (int j = 0; j < 4; ++j) acc[i][j] = (v8f)0.0f;

    // fragment addressing (ISA 7.12.2 layouts)
    const int ar  = lane & 15;                 // A row within 16x32 tile
    const int akb = (lane >> 4) ? 8 : 0;       // A k-base: lanes 0-15 -> K 0..7/16..23
    const int br  = lane & 15;                 // B column within tile
    const int bkb = (lane >> 4) ? 8 : 0;       // B k-base: lanes 0-15 -> K 0..7/16..23

    for (int k0 = 0; k0 < KDIM; k0 += 32) {
        // ---- stage 128x32 tiles of A and B into LDS (32 B per thread each)
        reinterpret_cast<uint4*>(asw)[0] = *reinterpret_cast<const uint4*>(xg + k0);
        reinterpret_cast<uint4*>(asw + 8)[0] = *reinterpret_cast<const uint4*>(xg + k0 + 8);
        reinterpret_cast<uint4*>(bsw)[0] = *reinterpret_cast<const uint4*>(wg + k0);
        reinterpret_cast<uint4*>(bsw + 8)[0] = *reinterpret_cast<const uint4*>(wg + k0 + 8);
        __syncthreads();

        // ---- A fragments (two 16-row tiles)
        v16bf afrag[2];
#pragma unroll
        for (int i = 0; i < 2; ++i) {
            const __bf16* base = As + (wm * 32 + i * 16 + ar) * LDS_STRIDE;
            uint4* p = reinterpret_cast<uint4*>(&afrag[i]);
            p[0] = *reinterpret_cast<const uint4*>(base + akb);        // K akb..akb+7
            p[1] = *reinterpret_cast<const uint4*>(base + akb + 16);   // K akb+16..akb+23
        }
        // ---- B fragments (four 16-col tiles)
        v16bf bfrag[4];
#pragma unroll
        for (int j = 0; j < 4; ++j) {
            const __bf16* base = Bs + (wn * 64 + j * 16 + br) * LDS_STRIDE + bkb;
            uint4* p = reinterpret_cast<uint4*>(&bfrag[j]);
            p[0] = *reinterpret_cast<const uint4*>(base);              // K bkb..bkb+7
            p[1] = *reinterpret_cast<const uint4*>(base + 16);         // K bkb+16..bkb+23
        }

#pragma unroll
        for (int i = 0; i < 2; ++i)
#pragma unroll
            for (int j = 0; j < 4; ++j)
                acc[i][j] = __builtin_amdgcn_wmma_f32_16x16x32_bf16(
                    /*neg_a=*/false, afrag[i], /*neg_b=*/false, bfrag[j],
                    /*c_mod=*/(short)0, acc[i][j],
                    /*reuse_a=*/false, /*reuse_b=*/false);
        __syncthreads();
    }

    // ---- epilogue: bias + ReLU + Dale sign, scatter per C/D layout
    const int colBase = bn * 128 + wn * 64;
    const int rowBase = bm * 128 + wm * 32;
    const int ln  = lane & 15;
    const int lh8 = (lane >> 4) << 3;   // +8 rows for lanes 16-31
#pragma unroll
    for (int j = 0; j < 4; ++j) {
        const int col = colBase + j * 16 + ln;
        const float bv  = bias[col];
        const float sgn = (col < EXC_DIM) ? 1.0f : -1.0f;
#pragma unroll
        for (int i = 0; i < 2; ++i) {
            const int row0 = rowBase + i * 16 + lh8;
#pragma unroll
            for (int v = 0; v < 8; ++v) {
                float val = fmaxf(acc[i][j][v] + bv, 0.0f) * sgn;
                out[(size_t)(row0 + v) * NDIM + col] = val;
            }
        }
    }
}

// ---------------------------------------------------------------------------
// Sequential GIF scan. One thread per (batch, neuron) chain; consecutive
// threads map to consecutive neurons -> fully coalesced (B,H) slabs per step.
// ---------------------------------------------------------------------------
__global__ __launch_bounds__(256)
void gif_scan(const float* __restrict__ iall,
              float* __restrict__ spikes,
              float* __restrict__ v_out,
              float* __restrict__ theta_out) {
    const int g = blockIdx.x * blockDim.x + threadIdx.x;   // 0 .. 32767
    const int b = g >> 11;            // batch
    const int h = g & (NDIM - 1);     // neuron
    const size_t base = (size_t)b * TSTEPS * NDIM + h;

    float v = 0.0f;
    float theta = 1.0f;               // THRESHOLD
    for (int t = 0; t < TSTEPS; ++t) {
        const size_t idx = base + (size_t)t * NDIM;
        float i_t = iall[idx];
        v = v * DECAY_C + i_t;
        float clampLim = 2.0f * L_LEVELS * theta;           // L*theta*2
        v = fminf(fmaxf(v, -clampLim), clampLim);
        float s = floorf(v / theta);
        s = fminf(fmaxf(s, 0.0f), L_LEVELS);
        v = v - s * theta;
        theta = theta + ALPHA_C * s - ALPHA_C * (theta - 1.0f);
        spikes[idx] = s;
    }
    v_out[g] = v;
    theta_out[g] = theta;
}

// ---------------------------------------------------------------------------
// launch
// ---------------------------------------------------------------------------
extern "C" void kernel_launch(void* const* d_in, const int* in_sizes, int n_in,
                              void* d_out, int out_size, void* d_ws, size_t ws_size,
                              hipStream_t stream) {
    const float* x     = (const float*)d_in[0];   // (16,1024,1024)
    const float* w_exc = (const float*)d_in[1];   // (1638,1024)
    const float* b_exc = (const float*)d_in[2];   // (1638,)
    const float* w_inh = (const float*)d_in[3];   // (410,1024)
    const float* b_inh = (const float*)d_in[4];   // (410,)
    float* out = (float*)d_out;

    // workspace layout
    char* ws = (char*)d_ws;
    __bf16* Xb = (__bf16*)ws;                                   // 33,554,432 B
    __bf16* Wb = (__bf16*)(ws + (size_t)33554432);              //  4,194,304 B
    float*  Bb = (float*)(ws + (size_t)33554432 + 4194304);     //      8,192 B
    float*  Ia = (float*)(ws + (size_t)33554432 + 4194304 + 8192); // 134 MB

    // d_out layout: spikes (16*1024*2048) | v_f (16*2048) | theta_f (16*2048)
    float* spikes   = out;
    float* v_f      = out + (size_t)MDIM * NDIM;
    float* theta_f  = v_f + (size_t)BATCH * NDIM;

    // phase 1: conversions
    {
        int n4 = (MDIM * KDIM) / 4;                  // 4,194,304
        cvt_x_bf16<<<n4 / 256, 256, 0, stream>>>(x, Xb, n4);
    }
    {
        int n4 = (NDIM * KDIM) / 4;                  // 524,288
        cvt_w_bf16<<<n4 / 256, 256, 0, stream>>>(w_exc, w_inh, Wb, n4);
    }
    pack_bias<<<NDIM / 256, 256, 0, stream>>>(b_exc, b_inh, Bb);

    // phase 2: WMMA GEMM + bias + ReLU + Dale
    dim3 gemmGrid(MDIM / 128, NDIM / 128);           // (128, 16)
    gemm_bias_dale<<<gemmGrid, 256, 0, stream>>>(Xb, Wb, Bb, Ia);

    // phase 3: scan
    gif_scan<<<(BATCH * NDIM) / 256, 256, 0, stream>>>(Ia, spikes, v_f, theta_f);
}
